// TimeFilm_79319456022770
// MI455X (gfx1250) — matrix operand order repaired and checked by reading, not compiled, not run on hardware
//
#include <hip/hip_runtime.h>
#include <hip/hip_bf16.h>

// ---------------------------------------------------------------------------
// TimeFilm on gfx1250 (MI455X): WMMA f16->f32 matrix path, M-register-blocked.
//   gamma_pre = [sin||cos] @ [a;b]   (K=512, N=256)
//   beta      = [sin||cos] @ [v;w]   (K=512, N=256)
//   z         = x*lp_w*tanh(gamma_pre) + beta
//   out       = z @ lp2_w^T          (K=256, N=256)
//
// Tiling: 64 tokens/block, 256 threads (8 wave32). Each wave owns 32 N cols
// (2 N-tiles of 16) and 4 M-tiles of 16 -> each B fragment fetched from L2 is
// reused by 4 WMMAs (4x less L2 traffic than a 16-token tile).
// ---------------------------------------------------------------------------

typedef __attribute__((ext_vector_type(16))) _Float16 v16h;
typedef __attribute__((ext_vector_type(8)))  _Float16 v8h;
typedef __attribute__((ext_vector_type(8)))  float    v8f;

union HFRAG { v16h v; v8h h8[2]; };

#define HARM 256          // H
#define EDIM 256          // E
#define K1   512          // sin||cos contraction depth
#define SC_STRIDE 520     // 512 + 8 halves pad -> conflict-free ds_load_b128
#define Z_STRIDE  264     // 256 + 8 halves pad
#define M_BLK 64          // tokens per block
#define THREADS 256       // 8 waves (wave32)

// Workspace layout (f16 halves):
//   [0      , 131072)  gammaBt : Bt[n][k], k<256 -> a[k][n], k>=256 -> b[k-256][n]
//   [131072 , 262144)  betaBt  : Bt[n][k], k<256 -> v[k][n], k>=256 -> w[k-256][n]
//   [262144 , 327680)  lp2h    : lp2_w[f][e] row-major (already the transposed B)

__global__ __launch_bounds__(256) void timefilm_prep_kernel(
    const float* __restrict__ a, const float* __restrict__ b,
    const float* __restrict__ w, const float* __restrict__ v,
    const float* __restrict__ lp2, _Float16* __restrict__ ws)
{
    int idx = blockIdx.x * 256 + threadIdx.x;   // 0 .. 196607
    if (idx < 131072) {
        int n = idx >> 9;          // E index
        int k = idx & 511;         // stacked K index
        float g  = (k < HARM) ? a[k * EDIM + n] : b[(k - HARM) * EDIM + n];
        float bt = (k < HARM) ? v[k * EDIM + n] : w[(k - HARM) * EDIM + n];
        ws[idx]          = (_Float16)g;
        ws[131072 + idx] = (_Float16)bt;
    } else {
        int j = idx - 131072;      // 0 .. 65535
        ws[262144 + j] = (_Float16)lp2[j];
    }
}

__global__ __launch_bounds__(THREADS) void timefilm_wmma_kernel(
    const float* __restrict__ x, const float* __restrict__ t,
    const float* __restrict__ lp_w, const float* __restrict__ n_,
    const _Float16* __restrict__ gammaBt, const _Float16* __restrict__ betaBt,
    const _Float16* __restrict__ lp2h, float* __restrict__ out)
{
    // sin||cos tile (64 x 520 halves, 65KB); the z tile (64 x 264) aliases it
    // after stage 2 finishes reading.
    __shared__ _Float16 smem[M_BLK * SC_STRIDE];
    __shared__ float ts[M_BLK];
    __shared__ float xs[M_BLK];
    _Float16* const sc = smem;
    _Float16* const zt = smem;

    const int tid     = threadIdx.x;
    const int lane    = tid & 31;
    const int wave    = tid >> 5;          // 0..7, each owns 32 N columns
    const int lmod    = lane & 15;
    const int halfSel = lane >> 4;         // 0: lanes 0-15, 1: lanes 16-31
    const int mBase   = halfSel * 8;       // C/D row base per ISA layout
    const size_t tokBase = (size_t)blockIdx.x * M_BLK;

    if (tid < M_BLK) {
        ts[tid] = t[tokBase + tid];
        xs[tid] = x[tokBase + tid];
    }
    __syncthreads();

    // ---- Stage 1: harmonics tile (64 tokens x 512) into LDS as f16 --------
    // 256 threads -> h == tid; the frequency is register-resident.
    const float TWO_PI = 6.28318530717958647692f;
    const float nf = TWO_PI * n_[tid];
#pragma unroll 8
    for (int m = 0; m < M_BLK; ++m) {
        float ang = ts[m] * nf;
        float s = __sinf(ang);
        float c = __cosf(ang);
        sc[m * SC_STRIDE + tid]        = (_Float16)s;   // K = h
        sc[m * SC_STRIDE + HARM + tid] = (_Float16)c;   // K = 256 + h
    }
    __syncthreads();

    // ---- Stage 2: gamma/beta GEMM, K=512 ----------------------------------
    v8f accG[4][2] = {};   // [mt][nt]
    v8f accB[4][2] = {};

    // A-fragment source: contiguous k order, same as B. lane<16: K {0..15};
    // lane>=16: K {16..31} of row (lane&15).
    const _Float16* aRow = &sc[(size_t)lmod * SC_STRIDE + halfSel * 16];

#pragma unroll
    for (int kt = 0; kt < K1 / 32; ++kt) {
        const int kb = kt * 32;
        HFRAG af[4];
#pragma unroll
        for (int mt = 0; mt < 4; ++mt) {
            const _Float16* ap = aRow + (size_t)(mt * 16) * SC_STRIDE + kb;
            af[mt].h8[0] = *(const v8h*)(ap);
            af[mt].h8[1] = *(const v8h*)(ap + 8);
        }
#pragma unroll
        for (int nt = 0; nt < 2; ++nt) {
            const int nG = wave * 32 + nt * 16 + lmod;
            // B-fragment: lane half selects K 0-15 / 16-31 of the 32-K tile;
            // Bt is column-major so this is 16 contiguous halves.
            const _Float16* bg = gammaBt + (size_t)nG * K1 + kb + halfSel * 16;
            const _Float16* bb = betaBt  + (size_t)nG * K1 + kb + halfSel * 16;
            HFRAG bgf, bbf;
            bgf.h8[0] = *(const v8h*)(bg);
            bgf.h8[1] = *(const v8h*)(bg + 8);
            bbf.h8[0] = *(const v8h*)(bb);
            bbf.h8[1] = *(const v8h*)(bb + 8);
#pragma unroll
            for (int mt = 0; mt < 4; ++mt) {
                accG[mt][nt] = __builtin_amdgcn_wmma_f32_16x16x32_f16(
                    false, af[mt].v, false, bgf.v, (short)0, accG[mt][nt],
                    false, false);
                accB[mt][nt] = __builtin_amdgcn_wmma_f32_16x16x32_f16(
                    false, af[mt].v, false, bbf.v, (short)0, accB[mt][nt],
                    false, false);
            }
        }
    }
    __syncthreads();   // all sc reads done; zt may now alias the same LDS

    // ---- Stage 3: FiLM epilogue -> z tile (f16, LDS) ----------------------
#pragma unroll
    for (int nt = 0; nt < 2; ++nt) {
        const int col = wave * 32 + nt * 16 + lmod;
        const float lpw = lp_w[col];
#pragma unroll
        for (int mt = 0; mt < 4; ++mt) {
#pragma unroll
            for (int r = 0; r < 8; ++r) {
                const int m = mt * 16 + mBase + r;
                const float g = accG[mt][nt][r];
                // NaN-safe tanh: 1 - 2/(exp(2g)+1)  (v_exp_f32 fast path)
                const float e = __expf(2.0f * g);
                const float gamma = 1.0f - 2.0f / (e + 1.0f);
                const float zv = xs[m] * lpw * gamma + accB[mt][nt][r];
                zt[m * Z_STRIDE + col] = (_Float16)zv;
            }
        }
    }
    __syncthreads();

    // ---- Stage 4: output projection, K=256 --------------------------------
    v8f acc2[4][2] = {};
    const _Float16* zRow = &zt[(size_t)lmod * Z_STRIDE + halfSel * 16];
#pragma unroll
    for (int kt = 0; kt < EDIM / 32; ++kt) {
        const int kb = kt * 32;
        HFRAG af[4];
#pragma unroll
        for (int mt = 0; mt < 4; ++mt) {
            const _Float16* ap = zRow + (size_t)(mt * 16) * Z_STRIDE + kb;
            af[mt].h8[0] = *(const v8h*)(ap);
            af[mt].h8[1] = *(const v8h*)(ap + 8);
        }
#pragma unroll
        for (int nt = 0; nt < 2; ++nt) {
            const int nG = wave * 32 + nt * 16 + lmod;
            const _Float16* bp = lp2h + (size_t)nG * EDIM + kb + halfSel * 16;
            HFRAG bf;
            bf.h8[0] = *(const v8h*)(bp);
            bf.h8[1] = *(const v8h*)(bp + 8);
#pragma unroll
            for (int mt = 0; mt < 4; ++mt) {
                acc2[mt][nt] = __builtin_amdgcn_wmma_f32_16x16x32_f16(
                    false, af[mt].v, false, bf.v, (short)0, acc2[mt][nt],
                    false, false);
            }
        }
    }

    // ---- Stage 5: store f32 output ---------------------------------------
#pragma unroll
    for (int nt = 0; nt < 2; ++nt) {
        const int col = wave * 32 + nt * 16 + lmod;
#pragma unroll
        for (int mt = 0; mt < 4; ++mt) {
#pragma unroll
            for (int r = 0; r < 8; ++r) {
                const int m = mt * 16 + mBase + r;
                out[(tokBase + m) * EDIM + col] = acc2[mt][nt][r];
            }
        }
    }
}

extern "C" void kernel_launch(void* const* d_in, const int* in_sizes, int n_in,
                              void* d_out, int out_size, void* d_ws, size_t ws_size,
                              hipStream_t stream) {
    (void)in_sizes; (void)n_in; (void)out_size; (void)ws_size;
    const float* x     = (const float*)d_in[0];   // [B,S]
    const float* t     = (const float*)d_in[1];   // [B,S]
    // d_in[2] = band_info (int64) -- unused by the reference computation
    const float* a     = (const float*)d_in[3];   // [H,E]
    const float* b     = (const float*)d_in[4];   // [H,E]
    const float* w     = (const float*)d_in[5];   // [H,E]  (dict order: w before v)
    const float* v     = (const float*)d_in[6];   // [H,E]
    const float* lp_w  = (const float*)d_in[7];   // [E]
    const float* lp2_w = (const float*)d_in[8];   // [E,E]
    const float* n_    = (const float*)d_in[9];   // [H]

    _Float16* ws  = (_Float16*)d_ws;
    float*    out = (float*)d_out;

    // One-time (per call) weight conversion + transpose into f16 workspace.
    timefilm_prep_kernel<<<768, 256, 0, stream>>>(a, b, w, v, lp2_w, ws);

    // 131072 tokens / 64 per block.
    timefilm_wmma_kernel<<<2048, THREADS, 0, stream>>>(
        x, t, lp_w, n_, ws, ws + 131072, ws + 262144, out);
}
